// MultiHeadAttention_37778532336435
// MI455X (gfx1250) — matrix operand order repaired and checked by reading, not compiled, not run on hardware
//
#include <hip/hip_runtime.h>
#include <hip/hip_bf16.h>
#include <math.h>

// MHA forward for MI455X (gfx1250, wave32, WMMA).
// Shapes: B=2, S=2048, H=1024, NH=16, DK=64.

typedef _Float16 half_t;
typedef __attribute__((ext_vector_type(16))) _Float16 v16h;
typedef __attribute__((ext_vector_type(8)))  float    v8f;
typedef __attribute__((ext_vector_type(4)))  int      i32x4;

#define BD   2
#define SD   2048
#define HDIM 1024
#define NHD  16
#define DKD  64
#define KDIM 1024   // inner dim of all projections == HDIM

#define AS1 __attribute__((address_space(1)))
#define AS3 __attribute__((address_space(3)))

#if __has_builtin(__builtin_amdgcn_global_load_async_to_lds_b128) && \
    __has_builtin(__builtin_amdgcn_s_wait_asynccnt)
#define HAVE_ASYNC_LDS 1
#else
#define HAVE_ASYNC_LDS 0
#endif

// 16-byte global -> LDS copy; async (ASYNCcnt-tracked) when available.
__device__ __forceinline__ void cp16_g2l(const half_t* g, half_t* l) {
#if HAVE_ASYNC_LDS
  __builtin_amdgcn_global_load_async_to_lds_b128(
      (AS1 i32x4*)g, (AS3 i32x4*)l, 0, 0);
#else
  *(uint4*)l = *(const uint4*)g;
#endif
}
__device__ __forceinline__ void async_wait() {
#if HAVE_ASYNC_LDS
  __builtin_amdgcn_s_wait_asynccnt(0);
#endif
}

// Load a 16-half WMMA fragment as two 16-byte chunks.
__device__ __forceinline__ v16h frag_ld(const half_t* p0, const half_t* p1) {
  v16h r;
  ((uint4*)&r)[0] = *(const uint4*)p0;
  ((uint4*)&r)[1] = *(const uint4*)p1;
  return r;
}

__device__ __forceinline__ v8f wmma_f16(v16h a, v16h b, v8f c) {
  return __builtin_amdgcn_wmma_f32_16x16x32_f16(
      false, a, false, b, (short)0, c, false, false);
}

// ---------------------------------------------------------------------------
// Tiled GEMM: Out[m,n] = sum_k A[m,k] * W[n,k] + bias[n]   (A @ W^T + b)
// Block tile 128x128, 8 waves as 4(M) x 2(N), K staged 64-deep in LDS
// (two WMMA K-steps / 16 WMMAs per barrier pair).
// A_IS_F16: A is f16 in workspace (ctx) -> async-to-LDS staging,
//           else fp32 (converted during staging).
// MODE 0: write f16, scattered into head layout [B,NH,S,DK] (QKV projections)
// MODE 1: write fp32 row-major [M,H] to d_out (output projection)
// ---------------------------------------------------------------------------
template<bool A_IS_F16, int MODE>
__global__ __launch_bounds__(256) void gemm128(
    const void* __restrict__ Aim, const float* __restrict__ Wf,
    const float* __restrict__ bias, half_t* __restrict__ out16,
    float* __restrict__ out32) {
  __shared__ __align__(16) half_t As[128][64];
  __shared__ __align__(16) half_t Bs[128][64];

  const int t    = threadIdx.x;
  const int wave = t >> 5;
  const int lane = t & 31;
  const int hlf  = lane >> 4;   // which 16-lane half of the wave
  const int l16  = lane & 15;
  const int wm   = (wave & 3) * 32;   // wave M offset inside block tile
  const int wn   = (wave >> 2) * 64;  // wave N offset inside block tile
  const int blockM = blockIdx.x * 128;
  const int blockN = blockIdx.y * 128;

  v8f acc[2][4];
#pragma unroll
  for (int mi = 0; mi < 2; ++mi)
#pragma unroll
    for (int ni = 0; ni < 4; ++ni) { v8f z = {}; acc[mi][ni] = z; }

  const int srow = t >> 1;           // 0..127  (tile row staged by this thread)
  const int scol = (t & 1) * 32;     // 0 or 32 (32 elements along K)

  for (int k0 = 0; k0 < KDIM; k0 += 64) {
    // ---- stage A tile ----
    if (A_IS_F16) {
      const half_t* ap = (const half_t*)Aim + (size_t)(blockM + srow) * KDIM + k0 + scol;
#pragma unroll
      for (int c = 0; c < 4; ++c) cp16_g2l(ap + 8 * c, &As[srow][scol + 8 * c]);
    } else {
      const float* ap = (const float*)Aim + (size_t)(blockM + srow) * KDIM + k0 + scol;
#pragma unroll
      for (int i = 0; i < 32; ++i) As[srow][scol + i] = (half_t)ap[i];
    }
    // ---- stage B tile: Bs[n][k] = W[blockN+n][k0+k] (W row n is B column n) ----
    {
      const float* wp = Wf + (size_t)(blockN + srow) * KDIM + k0 + scol;
#pragma unroll
      for (int i = 0; i < 32; ++i) Bs[srow][scol + i] = (half_t)wp[i];
    }
    async_wait();
    __syncthreads();

    // ---- two WMMA K-steps per stage ----
#pragma unroll
    for (int kk = 0; kk < 2; ++kk) {
      v16h afrag[2], bfrag[4];
#pragma unroll
      for (int mi = 0; mi < 2; ++mi) {
        const half_t* ar = &As[wm + mi * 16 + l16][kk * 32];
        afrag[mi] = frag_ld(ar + hlf * 8, ar + 16 + hlf * 8);
      }
#pragma unroll
      for (int ni = 0; ni < 4; ++ni) {
        const half_t* br = &Bs[wn + ni * 16 + l16][kk * 32];
        bfrag[ni] = frag_ld(br + hlf * 8, br + 16 + hlf * 8);
      }
#pragma unroll
      for (int mi = 0; mi < 2; ++mi)
#pragma unroll
        for (int ni = 0; ni < 4; ++ni)
          acc[mi][ni] = wmma_f16(afrag[mi], bfrag[ni], acc[mi][ni]);
    }
    __syncthreads();
  }

  // ---- epilogue: C layout is lane -> N = l16, VGPR r -> M = r + 8*hlf ----
#pragma unroll
  for (int mi = 0; mi < 2; ++mi) {
#pragma unroll
    for (int ni = 0; ni < 4; ++ni) {
      const int n  = blockN + wn + ni * 16 + l16;
      const float bv = bias[n];
#pragma unroll
      for (int r = 0; r < 8; ++r) {
        const int m = blockM + wm + mi * 16 + r + 8 * hlf;
        const float val = acc[mi][ni][r] + bv;
        if (MODE == 0) {
          const int b = m / SD, s = m % SD;
          const int h = n / DKD, d = n % DKD;
          out16[(((size_t)b * NHD + h) * SD + s) * DKD + d] = (half_t)val;
        } else {
          out32[(size_t)m * HDIM + n] = val;
        }
      }
    }
  }
}

// ---------------------------------------------------------------------------
// Flash attention: block = 128 query rows for one (b,h); each wave owns 16
// query rows (Q fragments resident in registers). Streams 64-row K/V tiles
// through LDS (K via async-to-LDS), online softmax, P bounced via LDS.
// ---------------------------------------------------------------------------
__global__ __launch_bounds__(256) void flash_attn(
    const half_t* __restrict__ qh, const half_t* __restrict__ kh,
    const half_t* __restrict__ vh, half_t* __restrict__ ctx) {
  __shared__ __align__(16) half_t LK[64][64];       // K tile  [krow][d]
  __shared__ __align__(16) half_t LV[64][64];       // V tile transposed [d][krow]
  __shared__ __align__(16) half_t LP[8][16][64];    // per-wave P tile [m][kcol]

  const int t    = threadIdx.x;
  const int wave = t >> 5;
  const int lane = t & 31;
  const int hlf  = lane >> 4;
  const int l16  = lane & 15;

  const int bh = blockIdx.y;           // b * NH + h
  const int b  = bh / NHD;
  const int h  = bh % NHD;
  const int q0 = blockIdx.x * 128 + wave * 16;   // first query row of this wave

  const half_t* qbase = qh + (size_t)bh * SD * DKD;
  const half_t* kbase = kh + (size_t)bh * SD * DKD;
  const half_t* vbase = vh + (size_t)bh * SD * DKD;

  // Q A-fragments (16 rows x 64 d = 2 K-steps), resident for the whole kernel.
  const half_t* qr = qbase + (size_t)(q0 + l16) * DKD;
  v16h qa0 = frag_ld(qr + hlf * 8,      qr + 16 + hlf * 8);
  v16h qa1 = frag_ld(qr + 32 + hlf * 8, qr + 48 + hlf * 8);

  float mrow[8], lrow[8];
  v8f O[4];
#pragma unroll
  for (int r = 0; r < 8; ++r) { mrow[r] = -1e30f; lrow[r] = 0.0f; }
#pragma unroll
  for (int nt = 0; nt < 4; ++nt) { v8f z = {}; O[nt] = z; }

  const int srow = t >> 2;           // 0..63 : staged K/V row
  const int scol = (t & 3) * 16;     // 0,16,32,48 along d
  const float scale = 0.125f;        // 1/sqrt(DK)

  for (int kt = 0; kt < SD / 64; ++kt) {
    // ---- stage K tile (async global->LDS) and V tile (transposed, manual) ----
    const half_t* krp = kbase + (size_t)(kt * 64 + srow) * DKD + scol;
    cp16_g2l(krp,     &LK[srow][scol]);
    cp16_g2l(krp + 8, &LK[srow][scol + 8]);
    const half_t* vrp = vbase + (size_t)(kt * 64 + srow) * DKD + scol;
#pragma unroll
    for (int j = 0; j < 16; ++j) LV[scol + j][srow] = vrp[j];
    if (kt + 1 < SD / 64) {   // hint next tiles into cache (global_prefetch_b8)
      __builtin_prefetch(kbase + (size_t)((kt + 1) * 64 + srow) * DKD + scol, 0, 1);
      __builtin_prefetch(vbase + (size_t)((kt + 1) * 64 + srow) * DKD + scol, 0, 1);
    }
    async_wait();
    __syncthreads();

    // ---- scores: S = Q @ K^T  (B column n = K row n, contiguous in d) ----
    v8f sc[4];
#pragma unroll
    for (int nt = 0; nt < 4; ++nt) {
      v8f a = {};
      const half_t* br = &LK[nt * 16 + l16][0];
      a = wmma_f16(qa0, frag_ld(br + hlf * 8,      br + 16 + hlf * 8),      a);
      a = wmma_f16(qa1, frag_ld(br + 32 + hlf * 8, br + 48 + hlf * 8), a);
      sc[nt] = a;
    }

    // ---- online softmax (row = r + 8*hlf lives in lanes sharing l16 group) ----
#pragma unroll
    for (int r = 0; r < 8; ++r) {
      float sv[4];
#pragma unroll
      for (int nt = 0; nt < 4; ++nt) sv[nt] = sc[nt][r] * scale;
      float mx = fmaxf(fmaxf(sv[0], sv[1]), fmaxf(sv[2], sv[3]));
#pragma unroll
      for (int off = 1; off < 16; off <<= 1)
        mx = fmaxf(mx, __shfl_xor(mx, off, 32));   // xor<16 stays in half-wave
      const float mnew = fmaxf(mrow[r], mx);
      const float corr = __expf(mrow[r] - mnew);
      mrow[r] = mnew;
      float rs = 0.0f;
#pragma unroll
      for (int nt = 0; nt < 4; ++nt) {
        const float p = __expf(sv[nt] - mnew);
        LP[wave][r + 8 * hlf][nt * 16 + l16] = (half_t)p;  // C-layout -> LDS
        rs += p;
      }
#pragma unroll
      for (int off = 1; off < 16; off <<= 1)
        rs += __shfl_xor(rs, off, 32);
      lrow[r] = lrow[r] * corr + rs;
#pragma unroll
      for (int nt = 0; nt < 4; ++nt) O[nt][r] *= corr;
    }
    __syncthreads();

    // ---- O += P @ V : A-frags of P from LDS, B column n = d (LV row d) ----
    const half_t* pr = &LP[wave][l16][0];
    v16h pa0 = frag_ld(pr + hlf * 8,      pr + 16 + hlf * 8);
    v16h pa1 = frag_ld(pr + 32 + hlf * 8, pr + 48 + hlf * 8);
#pragma unroll
    for (int nt = 0; nt < 4; ++nt) {
      const half_t* br = &LV[nt * 16 + l16][0];
      O[nt] = wmma_f16(pa0, frag_ld(br + hlf * 8,      br + 16 + hlf * 8),      O[nt]);
      O[nt] = wmma_f16(pa1, frag_ld(br + 32 + hlf * 8, br + 48 + hlf * 8), O[nt]);
    }
    __syncthreads();
  }

  // ---- finalize: ctx[b, s, h*DK + d] as f16 for the output projection ----
#pragma unroll
  for (int nt = 0; nt < 4; ++nt) {
#pragma unroll
    for (int r = 0; r < 8; ++r) {
      const int s = q0 + r + 8 * hlf;
      const float val = O[nt][r] / lrow[r];
      ctx[((size_t)b * SD + s) * HDIM + h * DKD + nt * 16 + l16] = (half_t)val;
    }
  }
}

// ---------------------------------------------------------------------------
extern "C" void kernel_launch(void* const* d_in, const int* in_sizes, int n_in,
                              void* d_out, int out_size, void* d_ws, size_t ws_size,
                              hipStream_t stream) {
  // setup_inputs() order: q, v, k, Wq, bq, Wk, bk, Wv, bv, Wo, bo  (note v before k)
  const float* q  = (const float*)d_in[0];
  const float* v  = (const float*)d_in[1];
  const float* k  = (const float*)d_in[2];
  const float* Wq = (const float*)d_in[3];
  const float* bq = (const float*)d_in[4];
  const float* Wk = (const float*)d_in[5];
  const float* bk = (const float*)d_in[6];
  const float* Wv = (const float*)d_in[7];
  const float* bv = (const float*)d_in[8];
  const float* Wo = (const float*)d_in[9];
  const float* bo = (const float*)d_in[10];
  float* out = (float*)d_out;

  const size_t elems = (size_t)BD * SD * HDIM;   // 4M elements
  half_t* qh  = (half_t*)d_ws;                   // [B,NH,S,DK] f16
  half_t* kh  = qh + elems;
  half_t* vh  = kh + elems;
  half_t* ctx = vh + elems;                      // [B,S,H] f16

  dim3 blk(256);
  dim3 ggrid((BD * SD) / 128, HDIM / 128);       // (32, 8)

  gemm128<false, 0><<<ggrid, blk, 0, stream>>>(q, Wq, bq, qh, nullptr);
  gemm128<false, 0><<<ggrid, blk, 0, stream>>>(k, Wk, bk, kh, nullptr);
  gemm128<false, 0><<<ggrid, blk, 0, stream>>>(v, Wv, bv, vh, nullptr);

  flash_attn<<<dim3(SD / 128, BD * NHD), blk, 0, stream>>>(qh, kh, vh, ctx);

  gemm128<true, 1><<<ggrid, blk, 0, stream>>>(ctx, Wo, bo, nullptr, out);
}
